// GraphConvClf_67327907332508
// MI455X (gfx1250) — matrix operand order repaired and checked by reading, not compiled, not run on hardware
//
#include <hip/hip_runtime.h>
#include <hip/hip_bf16.h>

// ---------------------------------------------------------------------------
// GraphConvClf on gfx1250: WMMA f16 GEMMs + fp32 atomic scatter + BN + pool.
// ---------------------------------------------------------------------------

#define NV 65536      // vertices
#define NE 196608     // edges
#define NM 64         // meshes
#define D1 256
#define D2 512
#define D3 1024
#define NH 128        // fc1 out

typedef __attribute__((ext_vector_type(16))) _Float16 v16h;
typedef __attribute__((ext_vector_type(8)))  _Float16 v8h;
typedef __attribute__((ext_vector_type(8)))  float    v8f;

// ---------------------------------------------------------------------------
// Layer 0: din=3, too small for WMMA. h0 = x@w0+b0, h1 = x@w1+b1, fp32 VALU.
// ---------------------------------------------------------------------------
__global__ __launch_bounds__(256) void layer0_kernel(
    const float* __restrict__ verts,
    const float* __restrict__ w0, const float* __restrict__ b0,
    const float* __restrict__ w1, const float* __restrict__ b1,
    float* __restrict__ X, float* __restrict__ Y)
{
    size_t t = (size_t)blockIdx.x * 256 + threadIdx.x;     // V*256 threads
    int    c = (int)(t & (D1 - 1));
    size_t r = t >> 8;
    float x0 = verts[r * 3 + 0];
    float x1 = verts[r * 3 + 1];
    float x2 = verts[r * 3 + 2];
    X[t] = x0 * w0[c] + x1 * w0[D1 + c] + x2 * w0[2 * D1 + c] + b0[c];
    Y[t] = x0 * w1[c] + x1 * w1[D1 + c] + x2 * w1[2 * D1 + c] + b1[c];
}

// ---------------------------------------------------------------------------
// Convert weight (K x N, f32, row-major) to transposed f16 (N x K) so WMMA
// B-fragments become contiguous 32B loads.
// ---------------------------------------------------------------------------
__global__ __launch_bounds__(256) void cvt_wt_kernel(
    const float* __restrict__ w, _Float16* __restrict__ wt, int K, int N)
{
    size_t t = (size_t)blockIdx.x * 256 + threadIdx.x;
    if (t >= (size_t)K * N) return;
    int k = (int)(t % K);
    int n = (int)(t / K);
    wt[t] = (_Float16)w[(size_t)k * N + n];
}

// ---------------------------------------------------------------------------
// WMMA GEMM: C[M,N] = A[M,K](f16) * Bt[N,K](f16)^T + bias, fp32 accum.
// Block = 256 threads = 8 waves (4 row-groups x 2 col-groups);
// block tile 128(M) x 128(N); wave tile 32x64 via 2x4 v_wmma_f32_16x16x32_f16
// (12 b128 loads per 8 WMMAs). K multiple of 32; N multiple of 128.
// ---------------------------------------------------------------------------
__global__ __launch_bounds__(256) void wmma_gemm_bias_kernel(
    const _Float16* __restrict__ A,    // M x K
    const _Float16* __restrict__ Bt,   // N x K (transposed weights)
    const float*    __restrict__ bias, // N
    float*          __restrict__ C,    // M x N
    int N, int K)
{
    const int lane = threadIdx.x & 31;
    const int wave = threadIdx.x >> 5;
    const int m0 = blockIdx.y * 128 + (wave >> 1) * 32;
    const int n0 = blockIdx.x * 128 + (wave & 1)  * 64;

    // A-fragment (16x32 f16): lane = row (0..15), lane-half picks
    // K ranges {0..15} vs {16..31} -> two contiguous v8h loads.
    const int aRow  = lane & 15;
    const int aHalf = (lane >> 4) << 4;           // 0 or 16 (halves)
    // B-fragment (32x16 f16): lane = column, 16 contiguous K values per lane.
    const int bCol  = lane & 15;
    const int bHalf = (lane >> 4) << 4;           // 0 or 16 (halves)

    const _Float16* ap[2];
    ap[0] = A + (size_t)(m0 + aRow) * K + aHalf;
    ap[1] = ap[0] + (size_t)16 * K;
    const _Float16* bp[4];
    bp[0] = Bt + (size_t)(n0 + bCol) * K + bHalf;
    bp[1] = bp[0] + (size_t)16 * K;
    bp[2] = bp[0] + (size_t)32 * K;
    bp[3] = bp[0] + (size_t)48 * K;

    v8f acc[2][4] = {};

    union AF { v16h v; v8h h[2]; };

    for (int kk = 0; kk < K; kk += 32) {
        AF a[2]; v16h b[4];
        #pragma unroll
        for (int i = 0; i < 2; ++i) {
            a[i].h[0] = *(const v8h*)(ap[i] + kk);
            a[i].h[1] = *(const v8h*)(ap[i] + kk + 8);
        }
        #pragma unroll
        for (int j = 0; j < 4; ++j)
            b[j] = *(const v16h*)(bp[j] + kk);

        if (kk + 32 < K) {                         // global_prefetch_b8
            __builtin_prefetch(ap[0] + kk + 32, 0, 1);
            __builtin_prefetch(ap[1] + kk + 32, 0, 1);
            __builtin_prefetch(bp[0] + kk + 32, 0, 1);
            __builtin_prefetch(bp[2] + kk + 32, 0, 1);
        }

        #pragma unroll
        for (int i = 0; i < 2; ++i)
            #pragma unroll
            for (int j = 0; j < 4; ++j)
                acc[i][j] = __builtin_amdgcn_wmma_f32_16x16x32_f16(
                    false, a[i].v, false, b[j], (short)0, acc[i][j],
                    false, false);
    }

    // C/D f32 layout: VGPR r -> (M = halfsel*8 + r, N = lane%16).
    const int cN = lane & 15;
    const int cM = (lane >> 4) << 3;
    #pragma unroll
    for (int i = 0; i < 2; ++i) {
        #pragma unroll
        for (int j = 0; j < 4; ++j) {
            int n = n0 + j * 16 + cN;
            float bv = bias[n];
            float* cp = C + (size_t)(m0 + i * 16 + cM) * N + n;
            #pragma unroll
            for (int r = 0; r < 8; ++r)
                cp[(size_t)r * N] = acc[i][j][r] + bv;
        }
    }
}

// ---------------------------------------------------------------------------
// Edge scatter: X[i] += Y[j]; X[j] += Y[i]  (fp32 global atomics).
// One wave per edge, lanes stripe the channel dim (coalesced atomic bursts).
// ---------------------------------------------------------------------------
__global__ __launch_bounds__(256) void scatter_kernel(
    const int* __restrict__ edges, const float* __restrict__ Y,
    float* __restrict__ X, int D)
{
    int wave = threadIdx.x >> 5;
    int lane = threadIdx.x & 31;
    int e = blockIdx.x * 8 + wave;
    if (e >= NE) return;
    int vi = edges[2 * e + 0];
    int vj = edges[2 * e + 1];
    const float* yi = Y + (size_t)vi * D;
    const float* yj = Y + (size_t)vj * D;
    float* xi = X + (size_t)vi * D;
    float* xj = X + (size_t)vj * D;
    for (int c = lane; c < D; c += 32) {
        atomicAdd(&xi[c], yj[c]);
        atomicAdd(&xj[c], yi[c]);
    }
}

// ---------------------------------------------------------------------------
// BN column stats: partial sums of x and x^2 per channel (atomics into f32).
// Block = 64 cols x 4 row-groups; coalesced 256B row reads.
// ---------------------------------------------------------------------------
__global__ __launch_bounds__(256) void bn_stats_kernel(
    const float* __restrict__ x, float* __restrict__ sum,
    float* __restrict__ sumsq, int D)
{
    int tx = threadIdx.x & 63;
    int ty = threadIdx.x >> 6;
    int c  = blockIdx.x * 64 + tx;
    int r0 = blockIdx.y * 256;
    float s = 0.f, s2 = 0.f;
    for (int r = r0 + ty; r < r0 + 256; r += 4) {
        float v = x[(size_t)r * D + c];
        s += v; s2 += v * v;
    }
    atomicAdd(&sum[c], s);
    atomicAdd(&sumsq[c], s2);
}

__global__ void bn_final_kernel(
    const float* __restrict__ sum, const float* __restrict__ sumsq,
    const float* __restrict__ g, const float* __restrict__ be,
    float* __restrict__ scale, float* __restrict__ shift, int D)
{
    int c = blockIdx.x * 256 + threadIdx.x;
    if (c >= D) return;
    const float invV = 1.0f / (float)NV;
    float mu  = sum[c] * invV;
    float var = sumsq[c] * invV - mu * mu;
    float rs  = rsqrtf(var + 1e-5f);
    float sc  = g[c] * rs;
    scale[c] = sc;
    shift[c] = be[c] - sc * mu;
}

// y = relu(x*scale+shift); write f16 copy (next GEMM input); optional f32.
__global__ __launch_bounds__(256) void bn_relu_kernel(
    float* __restrict__ x, const float* __restrict__ scale,
    const float* __restrict__ shift, _Float16* __restrict__ xh,
    int Dmask, int writeF32)
{
    size_t t = (size_t)blockIdx.x * 256 + threadIdx.x;
    int c = (int)(t & Dmask);
    float v = fmaxf(x[t] * scale[c] + shift[c], 0.0f);
    xh[t] = (_Float16)v;
    if (writeF32) x[t] = v;
}

// ---------------------------------------------------------------------------
// Segment pooling: counts + sums (rows are sorted by segment -> local runs).
// ---------------------------------------------------------------------------
__global__ __launch_bounds__(256) void seg_count_kernel(
    const int* __restrict__ idx, float* __restrict__ cnt)
{
    int t = blockIdx.x * 256 + threadIdx.x;
    if (t < NV) atomicAdd(&cnt[idx[t]], 1.0f);
}

__global__ __launch_bounds__(256) void seg_sum_kernel(
    const float* __restrict__ x, const int* __restrict__ idx,
    float* __restrict__ sums, int D)
{
    int c  = blockIdx.x * 256 + threadIdx.x;
    int r0 = blockIdx.y * 128;
    int cur = -1; float acc = 0.f;
    for (int r = r0; r < r0 + 128; ++r) {
        int s = idx[r];
        if (s != cur) {
            if (cur >= 0) atomicAdd(&sums[(size_t)cur * D + c], acc);
            cur = s; acc = 0.f;
        }
        acc += x[(size_t)r * D + c];
    }
    if (cur >= 0) atomicAdd(&sums[(size_t)cur * D + c], acc);
}

// h[m,t] = relu( sum_k (sums[m,k]/cnt[m]) * fc1_w[k,t] + fc1_b[t] )
__global__ __launch_bounds__(128) void fc1_kernel(
    const float* __restrict__ sums, const float* __restrict__ cnt,
    const float* __restrict__ w, const float* __restrict__ b,
    float* __restrict__ h)
{
    int m = blockIdx.x;
    int t = threadIdx.x;
    float inv = 1.0f / fmaxf(cnt[m], 1.0f);
    float acc = b[t];
    for (int k = 0; k < D3; ++k)
        acc += sums[(size_t)m * D3 + k] * inv * w[(size_t)k * NH + t];
    h[m * NH + t] = fmaxf(acc, 0.0f);
}

// 4 heads, outputs concatenated flat: style(64x3) sem(64x2) fun(64x4) aes(64x5)
__global__ __launch_bounds__(32) void heads_kernel(
    const float* __restrict__ h,
    const float* __restrict__ sw,  const float* __restrict__ sb,
    const float* __restrict__ mw,  const float* __restrict__ mb,
    const float* __restrict__ fw,  const float* __restrict__ fb,
    const float* __restrict__ aw,  const float* __restrict__ ab,
    float* __restrict__ out)
{
    int m = blockIdx.x;
    int t = threadIdx.x;
    if (t >= 14) return;
    int col, ch, off;
    const float *w, *b;
    if (t < 3)       { col = t;     ch = 3; off = 0;      w = sw; b = sb; }
    else if (t < 5)  { col = t - 3; ch = 2; off = NM * 3; w = mw; b = mb; }
    else if (t < 9)  { col = t - 5; ch = 4; off = NM * 5; w = fw; b = fb; }
    else             { col = t - 9; ch = 5; off = NM * 9; w = aw; b = ab; }
    float acc = b[col];
    for (int k = 0; k < NH; ++k)
        acc += h[m * NH + k] * w[k * ch + col];
    out[off + m * ch + col] = acc;
}

// ---------------------------------------------------------------------------
// Host-side launch
// ---------------------------------------------------------------------------
extern "C" void kernel_launch(void* const* d_in, const int* in_sizes, int n_in,
                              void* d_out, int out_size, void* d_ws, size_t ws_size,
                              hipStream_t stream) {
    (void)in_sizes; (void)n_in; (void)out_size; (void)ws_size;

    const float* verts    = (const float*)d_in[0];
    const int*   edges    = (const int*)  d_in[1];
    const int*   vidx     = (const int*)  d_in[2];
    const float* w0[3]    = {(const float*)d_in[3],  (const float*)d_in[9],  (const float*)d_in[15]};
    const float* b0[3]    = {(const float*)d_in[4],  (const float*)d_in[10], (const float*)d_in[16]};
    const float* w1[3]    = {(const float*)d_in[5],  (const float*)d_in[11], (const float*)d_in[17]};
    const float* b1[3]    = {(const float*)d_in[6],  (const float*)d_in[12], (const float*)d_in[18]};
    const float* gg[3]    = {(const float*)d_in[7],  (const float*)d_in[13], (const float*)d_in[19]};
    const float* be[3]    = {(const float*)d_in[8],  (const float*)d_in[14], (const float*)d_in[20]};
    const float* fc1_w    = (const float*)d_in[21];
    const float* fc1_b    = (const float*)d_in[22];
    const float* style_w  = (const float*)d_in[23];
    const float* style_b  = (const float*)d_in[24];
    const float* sem_w    = (const float*)d_in[25];
    const float* sem_b    = (const float*)d_in[26];
    const float* fun_w    = (const float*)d_in[27];
    const float* fun_b    = (const float*)d_in[28];
    const float* aes_w    = (const float*)d_in[29];
    const float* aes_b    = (const float*)d_in[30];
    float* out = (float*)d_out;

    // --- workspace layout ---
    char* w = (char*)d_ws;
    size_t off = 0;
    float*    X    = (float*)(w + off);    off += (size_t)NV * D3 * 4;   // h0 / nbr accum
    float*    Y    = (float*)(w + off);    off += (size_t)NV * D3 * 4;   // h1
    _Float16* AH   = (_Float16*)(w + off); off += (size_t)NV * D3 * 2;   // f16 activations
    _Float16* WT0  = (_Float16*)(w + off); off += (size_t)D2 * D1 * 2;   // w0_1^T
    _Float16* WT1  = (_Float16*)(w + off); off += (size_t)D2 * D1 * 2;   // w1_1^T
    _Float16* WT2  = (_Float16*)(w + off); off += (size_t)D3 * D2 * 2;   // w0_2^T
    _Float16* WT3  = (_Float16*)(w + off); off += (size_t)D3 * D2 * 2;   // w1_2^T
    float*    sum  = (float*)(w + off);    off += (size_t)D3 * 4;        // bn sum
    float*    ssq  = (float*)(w + off);    off += (size_t)D3 * 4;        // bn sumsq
    float*    scl  = (float*)(w + off);    off += (size_t)D3 * 4;
    float*    sft  = (float*)(w + off);    off += (size_t)D3 * 4;
    float*    psum = (float*)(w + off);    off += (size_t)NM * D3 * 4;   // pooled sums
    float*    cnt  = (float*)(w + off);    off += (size_t)NM * 4;        // pooled counts
    float*    hbuf = (float*)(w + off);    off += (size_t)NM * NH * 4;   // fc1 out

    // --- convert weights to f16 transposed (N x K) ---
    cvt_wt_kernel<<<(D1 * D2 + 255) / 256, 256, 0, stream>>>(w0[1], WT0, D1, D2);
    cvt_wt_kernel<<<(D1 * D2 + 255) / 256, 256, 0, stream>>>(w1[1], WT1, D1, D2);
    cvt_wt_kernel<<<(D2 * D3 + 255) / 256, 256, 0, stream>>>(w0[2], WT2, D2, D3);
    cvt_wt_kernel<<<(D2 * D3 + 255) / 256, 256, 0, stream>>>(w1[2], WT3, D2, D3);

    const int dims[3] = {D1, D2, D3};
    for (int L = 0; L < 3; ++L) {
        const int D = dims[L];
        if (L == 0) {
            layer0_kernel<<<(size_t)NV * D1 / 256, 256, 0, stream>>>(
                verts, w0[0], b0[0], w1[0], b1[0], X, Y);
        } else {
            const _Float16* WA = (L == 1) ? WT0 : WT2;
            const _Float16* WB = (L == 1) ? WT1 : WT3;
            const int K = dims[L - 1];
            dim3 grid(D / 128, NV / 128);
            wmma_gemm_bias_kernel<<<grid, 256, 0, stream>>>(AH, WA, b0[L], X, D, K);
            wmma_gemm_bias_kernel<<<grid, 256, 0, stream>>>(AH, WB, b1[L], Y, D, K);
        }
        // nbr scatter: X += scatter(Y)
        scatter_kernel<<<NE / 8, 256, 0, stream>>>(edges, Y, X, D);
        // batch-norm + relu
        hipMemsetAsync(sum, 0, (size_t)2 * D3 * 4, stream);   // sum + ssq (contiguous)
        dim3 sgrid(D / 64, NV / 256);
        bn_stats_kernel<<<sgrid, 256, 0, stream>>>(X, sum, ssq, D);
        bn_final_kernel<<<(D + 255) / 256, 256, 0, stream>>>(sum, ssq, gg[L], be[L], scl, sft, D);
        bn_relu_kernel<<<(size_t)NV * D / 256, 256, 0, stream>>>(
            X, scl, sft, AH, D - 1, (L == 2) ? 1 : 0);
    }

    // --- segment mean pool (64 meshes) ---
    hipMemsetAsync(psum, 0, (size_t)NM * D3 * 4 + NM * 4, stream); // psum + cnt contiguous
    seg_count_kernel<<<NV / 256, 256, 0, stream>>>(vidx, cnt);
    dim3 pgrid(D3 / 256, NV / 128);
    seg_sum_kernel<<<pgrid, 256, 0, stream>>>(X, vidx, psum, D3);

    // --- fc1 + heads ---
    fc1_kernel<<<NM, 128, 0, stream>>>(psum, cnt, fc1_w, fc1_b, hbuf);
    heads_kernel<<<NM, 32, 0, stream>>>(hbuf, style_w, style_b, sem_w, sem_b,
                                        fun_w, fun_b, aes_w, aes_b, out);
}
